// GNNLSTM_62053687492936
// MI455X (gfx1250) — compile-verified, matrix operand order checked by reading, not run on hardware
//
#include <hip/hip_runtime.h>
#include <math.h>

#define N_NODES 2048
#define N_GRAPHS 64
#define N_EDGES 65536
#define F_IN 7680
#define H1 5000
#define SL 2500

typedef __attribute__((ext_vector_type(16))) __bf16 v16bf;
typedef __attribute__((ext_vector_type(8)))  __bf16 v8bf;
typedef __attribute__((ext_vector_type(4)))  __bf16 v4bf;
typedef __attribute__((ext_vector_type(8)))  float  v8f;

#define WMMA_BF16(A, B, C)                                                    \
  __builtin_amdgcn_wmma_f32_16x16x32_bf16(false, (A), false, (B), (short)0,   \
                                          (C), false, false)

// ---------------------------------------------------------------------------
// Fragment loaders for V_WMMA_F32_16X16X32_BF16 (ISA 7.12.2 layout):
// row from lane&15, K offset from lane>>4 and VGPR index.
// f32 source: 4 contiguous float4 loads + v_cvt_pk_bf16_f32.
// bf16 source: 2 contiguous 16-byte loads, no conversion.
// ---------------------------------------------------------------------------
__device__ __forceinline__ v16bf load_frag(const float* __restrict__ row,
                                           int k0, int half) {
  const float4* p0 = (const float4*)(row + k0 + (half << 3));
  const float4* p1 = (const float4*)(row + k0 + 16 + (half << 3));
  float4 a = p0[0], b = p0[1], c = p1[0], d = p1[1];
  v16bf f;
  f[0] = (__bf16)a.x;  f[1] = (__bf16)a.y;  f[2]  = (__bf16)a.z;  f[3]  = (__bf16)a.w;
  f[4] = (__bf16)b.x;  f[5] = (__bf16)b.y;  f[6]  = (__bf16)b.z;  f[7]  = (__bf16)b.w;
  f[8] = (__bf16)c.x;  f[9] = (__bf16)c.y;  f[10] = (__bf16)c.z;  f[11] = (__bf16)c.w;
  f[12] = (__bf16)d.x; f[13] = (__bf16)d.y; f[14] = (__bf16)d.z;  f[15] = (__bf16)d.w;
  return f;
}

__device__ __forceinline__ v16bf load_frag(const __bf16* __restrict__ row,
                                           int k0, int half) {
  v8bf lo = *(const v8bf*)(row + k0 + (half << 3));
  v8bf hi = *(const v8bf*)(row + k0 + 16 + (half << 3));
  v16bf f;
#pragma unroll
  for (int s = 0; s < 8; ++s) {
    f[s] = lo[s];
    f[8 + s] = hi[s];
  }
  return f;
}

__device__ __forceinline__ v16bf load_frag_guard(const float* __restrict__ row,
                                                 int k0, int Klim, int half) {
  v16bf f;
#pragma unroll
  for (int v = 0; v < 8; ++v) {
    int kl = ((v >> 2) << 4) + (half << 3) + ((v & 3) << 1);
    int k = k0 + kl;
    f[2 * v]     = (__bf16)((k     < Klim) ? row[k]     : 0.f);
    f[2 * v + 1] = (__bf16)((k + 1 < Klim) ? row[k + 1] : 0.f);
  }
  return f;
}

__device__ __forceinline__ v16bf load_frag_guard(const __bf16* __restrict__ row,
                                                 int k0, int Klim, int half) {
  v16bf f;
#pragma unroll
  for (int v = 0; v < 8; ++v) {
    int kl = ((v >> 2) << 4) + (half << 3) + ((v & 3) << 1);
    int k = k0 + kl;
    f[2 * v]     = (k     < Klim) ? row[k]     : (__bf16)0.f;
    f[2 * v + 1] = (k + 1 < Klim) ? row[k + 1] : (__bf16)0.f;
  }
  return f;
}

// ---------------------------------------------------------------------------
__global__ void zero_k(float* __restrict__ p, long n) {
  long i = (long)blockIdx.x * blockDim.x + threadIdx.x;
  if (i < n) p[i] = 0.f;
}

// f32 -> bf16, 4 elements/thread (float4 in, 8-byte bf16x4 out)
__global__ void cvt_bf16_k(const float* __restrict__ in,
                           __bf16* __restrict__ out, long n4) {
  long i = (long)blockIdx.x * blockDim.x + threadIdx.x;
  if (i < n4) {
    float4 v = ((const float4*)in)[i];
    v4bf o;
    o[0] = (__bf16)v.x;
    o[1] = (__bf16)v.y;
    o[2] = (__bf16)v.z;
    o[3] = (__bf16)v.w;
    ((v4bf*)out)[i] = o;
  }
}

// ---------------------------------------------------------------------------
// Edge scatter: agg[dst] += w_e * x[src].  x/agg are L2-resident (63 MB each),
// so per-element global_atomic_add_f32 stays on-chip.
// ---------------------------------------------------------------------------
__global__ __launch_bounds__(256) void edge_scatter_k(
    const float* __restrict__ x, const int* __restrict__ ei,
    const float* __restrict__ ea, float* __restrict__ agg) {
  int e = blockIdx.x;
  int src = ei[e];
  int dst = ei[N_EDGES + e];
  float w = ea[e];
  const float* xs = x + (size_t)src * F_IN;
  float* ad = agg + (size_t)dst * F_IN;
  for (int j = threadIdx.x; j < F_IN; j += blockDim.x)
    atomicAdd(&ad[j], w * xs[j]);
}

// ---------------------------------------------------------------------------
// h1 = relu(agg @ Wrel^T + brel + x @ Wroot^T)
// M=2048, N=5000, K=7680.  2Mx4N register blocking per wave (8 accumulators:
// each A-fragment feeds 4 WMMAs, each B-fragment feeds 2).  8 waves arranged
// 4M x 2N -> 128x128 block tile, so weight streams are re-read only
// 2048/128 = 16 times.  WT = float (inline cvt) or __bf16 (pre-converted).
// ---------------------------------------------------------------------------
template <typename WT>
__global__ __launch_bounds__(256) void gemm_h1_k(
    const float* __restrict__ agg, const float* __restrict__ x,
    const WT* __restrict__ wrel, const WT* __restrict__ wroot,
    const float* __restrict__ brel, float* __restrict__ h1) {
  const int Nb = (H1 + 127) / 128;  // 40
  int bm = blockIdx.x / Nb;
  int bn = blockIdx.x % Nb;
  int wave = threadIdx.x >> 5;
  int mw = wave >> 1;
  int nw = wave & 1;
  int lane = threadIdx.x & 31;
  int rsel = lane & 15;
  int half = lane >> 4;
  int m0 = bm * 128 + mw * 32;
  int n0 = bn * 128 + nw * 64;

  const float* arow0 = agg + (size_t)(m0 + rsel) * F_IN;
  const float* arow1 = agg + (size_t)(m0 + 16 + rsel) * F_IN;
  const float* xrow0 = x + (size_t)(m0 + rsel) * F_IN;
  const float* xrow1 = x + (size_t)(m0 + 16 + rsel) * F_IN;
  const WT* wrelr[4];
  const WT* wrootr[4];
#pragma unroll
  for (int j = 0; j < 4; ++j) {
    int wr = n0 + 16 * j + rsel;
    if (wr > H1 - 1) wr = H1 - 1;  // clamp; garbage cols masked at store
    wrelr[j] = wrel + (size_t)wr * F_IN;
    wrootr[j] = wroot + (size_t)wr * F_IN;
  }

  v8f acc[2][4] = {};
  for (int k0 = 0; k0 < F_IN; k0 += 32) {
    __builtin_prefetch(arow0 + k0 + 256, 0, 3);
    __builtin_prefetch(wrelr[0] + k0 + 256, 0, 3);
    v16bf a0 = load_frag(arow0, k0, half);
    v16bf a1 = load_frag(arow1, k0, half);
#pragma unroll
    for (int j = 0; j < 4; ++j) {
      v16bf b = load_frag(wrelr[j], k0, half);
      acc[0][j] = WMMA_BF16(a0, b, acc[0][j]);
      acc[1][j] = WMMA_BF16(a1, b, acc[1][j]);
    }
  }
  for (int k0 = 0; k0 < F_IN; k0 += 32) {
    __builtin_prefetch(xrow0 + k0 + 256, 0, 3);
    __builtin_prefetch(wrootr[0] + k0 + 256, 0, 3);
    v16bf a0 = load_frag(xrow0, k0, half);
    v16bf a1 = load_frag(xrow1, k0, half);
#pragma unroll
    for (int j = 0; j < 4; ++j) {
      v16bf b = load_frag(wrootr[j], k0, half);
      acc[0][j] = WMMA_BF16(a0, b, acc[0][j]);
      acc[1][j] = WMMA_BF16(a1, b, acc[1][j]);
    }
  }

#pragma unroll
  for (int j = 0; j < 4; ++j) {
    int col = n0 + 16 * j + rsel;
    if (col < H1) {
      float bias = brel[col];
#pragma unroll
      for (int i = 0; i < 2; ++i) {
#pragma unroll
        for (int r = 0; r < 8; ++r) {
          int row = m0 + i * 16 + (half << 3) + r;
          float v = acc[i][j][r] + bias;
          h1[(size_t)row * H1 + col] = v > 0.f ? v : 0.f;
        }
      }
    }
  }
}

// ---------------------------------------------------------------------------
// GatedAttention triple GEMM: wave owns 32Mx32N for THREE weight streams
// (12 accumulators); A-fragments reused 6x, B-fragments 2x.  Block tile
// 128M x 64N.  gated = tanh(.W)*sigmoid(.U); value = .val
// ---------------------------------------------------------------------------
template <typename WT>
__global__ __launch_bounds__(256) void gemm_ga_k(
    const float* __restrict__ h1, const WT* __restrict__ wW,
    const float* __restrict__ bW, const WT* __restrict__ wU,
    const float* __restrict__ bU, const WT* __restrict__ wVal,
    const float* __restrict__ bVal, float* __restrict__ gated,
    float* __restrict__ value) {
  const int Nb = (H1 + 63) / 64;  // 79
  int bm = blockIdx.x / Nb;
  int bn = blockIdx.x % Nb;
  int wave = threadIdx.x >> 5;
  int mw = wave >> 1;
  int nw = wave & 1;
  int lane = threadIdx.x & 31;
  int rsel = lane & 15;
  int half = lane >> 4;
  int m0 = bm * 128 + mw * 32;
  int n0 = bn * 64 + nw * 32;

  const float* arow0 = h1 + (size_t)(m0 + rsel) * H1;
  const float* arow1 = h1 + (size_t)(m0 + 16 + rsel) * H1;
  const WT* wWr[2];
  const WT* wUr[2];
  const WT* wVr[2];
#pragma unroll
  for (int j = 0; j < 2; ++j) {
    int wr = n0 + 16 * j + rsel;
    if (wr > H1 - 1) wr = H1 - 1;
    wWr[j] = wW + (size_t)wr * H1;
    wUr[j] = wU + (size_t)wr * H1;
    wVr[j] = wVal + (size_t)wr * H1;
  }

  v8f accW[2][2] = {}, accU[2][2] = {}, accV[2][2] = {};
  const int kfull = (H1 / 32) * 32;  // 4992
  for (int k0 = 0; k0 < kfull; k0 += 32) {
    __builtin_prefetch(arow0 + k0 + 256, 0, 3);
    __builtin_prefetch(wWr[0] + k0 + 256, 0, 3);
    v16bf a0 = load_frag(arow0, k0, half);
    v16bf a1 = load_frag(arow1, k0, half);
#pragma unroll
    for (int j = 0; j < 2; ++j) {
      v16bf b0 = load_frag(wWr[j], k0, half);
      accW[0][j] = WMMA_BF16(a0, b0, accW[0][j]);
      accW[1][j] = WMMA_BF16(a1, b0, accW[1][j]);
      v16bf b1 = load_frag(wUr[j], k0, half);
      accU[0][j] = WMMA_BF16(a0, b1, accU[0][j]);
      accU[1][j] = WMMA_BF16(a1, b1, accU[1][j]);
      v16bf b2 = load_frag(wVr[j], k0, half);
      accV[0][j] = WMMA_BF16(a0, b2, accV[0][j]);
      accV[1][j] = WMMA_BF16(a1, b2, accV[1][j]);
    }
  }
  {  // K tail (5000 % 32 == 8)
    v16bf a0 = load_frag_guard(arow0, kfull, H1, half);
    v16bf a1 = load_frag_guard(arow1, kfull, H1, half);
#pragma unroll
    for (int j = 0; j < 2; ++j) {
      v16bf b0 = load_frag_guard(wWr[j], kfull, H1, half);
      accW[0][j] = WMMA_BF16(a0, b0, accW[0][j]);
      accW[1][j] = WMMA_BF16(a1, b0, accW[1][j]);
      v16bf b1 = load_frag_guard(wUr[j], kfull, H1, half);
      accU[0][j] = WMMA_BF16(a0, b1, accU[0][j]);
      accU[1][j] = WMMA_BF16(a1, b1, accU[1][j]);
      v16bf b2 = load_frag_guard(wVr[j], kfull, H1, half);
      accV[0][j] = WMMA_BF16(a0, b2, accV[0][j]);
      accV[1][j] = WMMA_BF16(a1, b2, accV[1][j]);
    }
  }

#pragma unroll
  for (int j = 0; j < 2; ++j) {
    int col = n0 + 16 * j + rsel;
    if (col < H1) {
      float biasW = bW[col], biasU = bU[col], biasV = bVal[col];
#pragma unroll
      for (int i = 0; i < 2; ++i) {
#pragma unroll
        for (int r = 0; r < 8; ++r) {
          int row = m0 + i * 16 + (half << 3) + r;
          float hg = tanhf(accW[i][j][r] + biasW);
          float ug = 1.f / (1.f + expf(-(accU[i][j][r] + biasU)));
          gated[(size_t)row * H1 + col] = hg * ug;
          value[(size_t)row * H1 + col] = accV[i][j][r] + biasV;
        }
      }
    }
  }
}

// ---------------------------------------------------------------------------
// scores[m] = gated[m,:] . Vw + Vb
// ---------------------------------------------------------------------------
__global__ __launch_bounds__(256) void score_k(const float* __restrict__ gated,
                                               const float* __restrict__ Vw,
                                               const float* __restrict__ Vb,
                                               float* __restrict__ scores) {
  __shared__ float red[256];
  int m = blockIdx.x;
  const float* g = gated + (size_t)m * H1;
  float s = 0.f;
  for (int j = threadIdx.x; j < H1; j += 256) s += g[j] * Vw[j];
  red[threadIdx.x] = s;
  __syncthreads();
  for (int st = 128; st > 0; st >>= 1) {
    if (threadIdx.x < st) red[threadIdx.x] += red[threadIdx.x + st];
    __syncthreads();
  }
  if (threadIdx.x == 0) scores[m] = red[0] + Vb[0];
}

// ---------------------------------------------------------------------------
// Softmax over ALL 2048 nodes (axis=0 global, per the reference)
// ---------------------------------------------------------------------------
__global__ __launch_bounds__(256) void softmax_k(const float* __restrict__ scores,
                                                 float* __restrict__ alpha) {
  __shared__ float red[256];
  int tid = threadIdx.x;
  float m = -1e30f;
  for (int i = tid; i < N_NODES; i += 256) m = fmaxf(m, scores[i]);
  red[tid] = m;
  __syncthreads();
  for (int st = 128; st > 0; st >>= 1) {
    if (tid < st) red[tid] = fmaxf(red[tid], red[tid + st]);
    __syncthreads();
  }
  float mx = red[0];
  __syncthreads();
  float s = 0.f;
  for (int i = tid; i < N_NODES; i += 256) s += expf(scores[i] - mx);
  red[tid] = s;
  __syncthreads();
  for (int st = 128; st > 0; st >>= 1) {
    if (tid < st) red[tid] += red[tid + st];
    __syncthreads();
  }
  float inv = 1.f / red[0];
  for (int i = tid; i < N_NODES; i += 256) alpha[i] = expf(scores[i] - mx) * inv;
}

// ---------------------------------------------------------------------------
// Global add pool: pooled[batch[n]] += alpha[n] * value[n]
// ---------------------------------------------------------------------------
__global__ __launch_bounds__(256) void pool_k(const float* __restrict__ value,
                                              const float* __restrict__ alpha,
                                              const int* __restrict__ batch,
                                              float* __restrict__ pooled) {
  int n = blockIdx.x;
  float a = alpha[n];
  int g = batch[n];
  const float* v = value + (size_t)n * H1;
  float* p = pooled + (size_t)g * H1;
  for (int j = threadIdx.x; j < H1; j += 256) atomicAdd(&p[j], a * v[j]);
}

// ---------------------------------------------------------------------------
// rearrange pooled 'b (sl i) -> sl b i'  (i=2)
// ---------------------------------------------------------------------------
__global__ void build_seq0_k(const float* __restrict__ pooled,
                             float* __restrict__ seq0) {
  int idx = blockIdx.x * 256 + threadIdx.x;  // (t*64+b)*2+i
  if (idx >= SL * N_GRAPHS * 2) return;
  int i = idx & 1;
  int tb = idx >> 1;
  int b = tb & 63;
  int t = tb >> 6;
  seq0[idx] = pooled[(size_t)b * H1 + t * 2 + i];
}

// ---------------------------------------------------------------------------
// One LSTM direction: thread b holds h/c in registers, weights in LDS,
// sequentially walks T=2500 steps. Gate order i,f,g,o (PyTorch).
// seq: [T][64][in_dim]; out: [T][64][6] at column offset col_off.
// ---------------------------------------------------------------------------
__global__ __launch_bounds__(64) void lstm_scan_k(
    const float* __restrict__ seq, int in_dim, const float* __restrict__ Wih,
    const float* __restrict__ Whh, const float* __restrict__ bih,
    const float* __restrict__ bhh, float* __restrict__ out, int col_off,
    int reverse) {
  __shared__ float sWih[12 * 6], sWhh[12 * 3], sb[12];
  int tid = threadIdx.x;
  for (int i = tid; i < 12 * in_dim; i += blockDim.x) sWih[i] = Wih[i];
  for (int i = tid; i < 36; i += blockDim.x) sWhh[i] = Whh[i];
  for (int i = tid; i < 12; i += blockDim.x) sb[i] = bih[i] + bhh[i];
  __syncthreads();
  if (tid >= N_GRAPHS) return;

  float h[3] = {0.f, 0.f, 0.f}, c[3] = {0.f, 0.f, 0.f};
  for (int s = 0; s < SL; ++s) {
    int t = reverse ? (SL - 1 - s) : s;
    const float* xrow = seq + ((size_t)t * N_GRAPHS + tid) * in_dim;
    float g[12];
#pragma unroll
    for (int j = 0; j < 12; ++j) {
      float acc = sb[j];
      for (int i = 0; i < in_dim; ++i) acc += xrow[i] * sWih[j * in_dim + i];
#pragma unroll
      for (int kk = 0; kk < 3; ++kk) acc += h[kk] * sWhh[j * 3 + kk];
      g[j] = acc;
    }
#pragma unroll
    for (int kk = 0; kk < 3; ++kk) {
      float ig = 1.f / (1.f + expf(-g[kk]));
      float fg = 1.f / (1.f + expf(-g[3 + kk]));
      float gg = tanhf(g[6 + kk]);
      float og = 1.f / (1.f + expf(-g[9 + kk]));
      c[kk] = fg * c[kk] + ig * gg;
      h[kk] = og * tanhf(c[kk]);
    }
    float* orow = out + ((size_t)t * N_GRAPHS + tid) * 6 + col_off;
    orow[0] = h[0];
    orow[1] = h[1];
    orow[2] = h[2];
  }
}

// ---------------------------------------------------------------------------
// Head: out[g] = sum_{t,c} seq1[t][g][c] * mlp_w[t*6+c] + mlp_b
// ---------------------------------------------------------------------------
__global__ __launch_bounds__(256) void head_k(const float* __restrict__ seq1,
                                              const float* __restrict__ mlpw,
                                              const float* __restrict__ mlpb,
                                              float* __restrict__ outp) {
  __shared__ float red[256];
  int g = blockIdx.x;
  float s = 0.f;
  for (int idx = threadIdx.x; idx < SL * 6; idx += 256) {
    int t = idx / 6, cc = idx % 6;
    s += seq1[((size_t)t * N_GRAPHS + g) * 6 + cc] * mlpw[idx];
  }
  red[threadIdx.x] = s;
  __syncthreads();
  for (int st = 128; st > 0; st >>= 1) {
    if (threadIdx.x < st) red[threadIdx.x] += red[threadIdx.x + st];
    __syncthreads();
  }
  if (threadIdx.x == 0) outp[g] = red[0] + mlpb[0];
}

// ---------------------------------------------------------------------------
extern "C" void kernel_launch(void* const* d_in, const int* in_sizes, int n_in,
                              void* d_out, int out_size, void* d_ws,
                              size_t ws_size, hipStream_t stream) {
  (void)in_sizes; (void)n_in; (void)out_size;
  const float* x          = (const float*)d_in[0];
  const int*   ei         = (const int*)d_in[1];
  const float* ea         = (const float*)d_in[2];
  const int*   batch      = (const int*)d_in[3];
  const float* gc1_rel_w  = (const float*)d_in[4];
  const float* gc1_rel_b  = (const float*)d_in[5];
  const float* gc1_root_w = (const float*)d_in[6];
  const float* ga_W_w     = (const float*)d_in[7];
  const float* ga_W_b     = (const float*)d_in[8];
  const float* ga_U_w     = (const float*)d_in[9];
  const float* ga_U_b     = (const float*)d_in[10];
  const float* ga_V_w     = (const float*)d_in[11];
  const float* ga_V_b     = (const float*)d_in[12];
  const float* ga_val_w   = (const float*)d_in[13];
  const float* ga_val_b   = (const float*)d_in[14];
  const float* mlp_w      = (const float*)d_in[15];
  const float* mlp_b      = (const float*)d_in[16];
  const float* l0f_Wih = (const float*)d_in[17];
  const float* l0f_Whh = (const float*)d_in[18];
  const float* l0f_bih = (const float*)d_in[19];
  const float* l0f_bhh = (const float*)d_in[20];
  const float* l0b_Wih = (const float*)d_in[21];
  const float* l0b_Whh = (const float*)d_in[22];
  const float* l0b_bih = (const float*)d_in[23];
  const float* l0b_bhh = (const float*)d_in[24];
  const float* l1f_Wih = (const float*)d_in[25];
  const float* l1f_Whh = (const float*)d_in[26];
  const float* l1f_bih = (const float*)d_in[27];
  const float* l1f_bhh = (const float*)d_in[28];
  const float* l1b_Wih = (const float*)d_in[29];
  const float* l1b_Whh = (const float*)d_in[30];
  const float* l1b_bih = (const float*)d_in[31];
  const float* l1b_bhh = (const float*)d_in[32];

  // ---- f32 workspace layout (floats). Peak ~155 MB; agg reused as gated.
  float* ws = (float*)d_ws;
  const long AGG_N = (long)N_NODES * F_IN;  // 15,728,640
  const long H1_N  = (long)N_NODES * H1;    // 10,240,000
  float* agg    = ws;
  float* gated  = agg;  // agg dead after gemm_h1; reuse for gated
  float* h1     = agg + AGG_N;
  float* value  = h1 + H1_N;
  float* scores = value + H1_N;
  float* alpha  = scores + N_NODES;
  float* pooled = alpha + N_NODES;
  float* seq0   = pooled + (long)N_GRAPHS * H1;
  float* seqA   = seq0 + (long)SL * N_GRAPHS * 2;
  float* seqB   = seqA + (long)SL * N_GRAPHS * 6;
  const long F32_TOTAL = (seqB + (long)SL * N_GRAPHS * 6) - ws;  // floats

  // ---- optional bf16 weight cache appended after the f32 region
  const long WGC_N = (long)H1 * F_IN;  // 38,400,000 per gc1 weight
  const long WGA_N = (long)H1 * H1;    // 25,000,000 per GA weight
  __bf16* bfbase   = (__bf16*)(ws + F32_TOTAL);
  __bf16* wrel_bf  = bfbase;
  __bf16* wroot_bf = wrel_bf + WGC_N;
  __bf16* wW_bf    = wroot_bf + WGC_N;
  __bf16* wU_bf    = wW_bf + WGA_N;
  __bf16* wVal_bf  = wU_bf + WGA_N;
  size_t need_big =
      (size_t)F32_TOTAL * 4 + (size_t)(2 * WGC_N + 3 * WGA_N) * 2;
  bool big = ws_size >= need_big;

  const int h1_blocks = (N_NODES / 128) * ((H1 + 127) / 128);  // 16*40 = 640
  const int ga_blocks = (N_NODES / 128) * ((H1 + 63) / 64);    // 16*79 = 1264

  zero_k<<<(int)((AGG_N + 255) / 256), 256, 0, stream>>>(agg, AGG_N);
  edge_scatter_k<<<N_EDGES, 256, 0, stream>>>(x, ei, ea, agg);

  if (big) {
    cvt_bf16_k<<<(int)(WGC_N / 4 + 255) / 256, 256, 0, stream>>>(gc1_rel_w,
                                                                 wrel_bf,
                                                                 WGC_N / 4);
    cvt_bf16_k<<<(int)(WGC_N / 4 + 255) / 256, 256, 0, stream>>>(gc1_root_w,
                                                                 wroot_bf,
                                                                 WGC_N / 4);
    cvt_bf16_k<<<(int)(WGA_N / 4 + 255) / 256, 256, 0, stream>>>(ga_W_w, wW_bf,
                                                                 WGA_N / 4);
    cvt_bf16_k<<<(int)(WGA_N / 4 + 255) / 256, 256, 0, stream>>>(ga_U_w, wU_bf,
                                                                 WGA_N / 4);
    cvt_bf16_k<<<(int)(WGA_N / 4 + 255) / 256, 256, 0, stream>>>(ga_val_w,
                                                                 wVal_bf,
                                                                 WGA_N / 4);
    gemm_h1_k<__bf16><<<h1_blocks, 256, 0, stream>>>(agg, x, wrel_bf, wroot_bf,
                                                     gc1_rel_b, h1);
    gemm_ga_k<__bf16><<<ga_blocks, 256, 0, stream>>>(h1, wW_bf, ga_W_b, wU_bf,
                                                     ga_U_b, wVal_bf, ga_val_b,
                                                     gated, value);
  } else {
    gemm_h1_k<float><<<h1_blocks, 256, 0, stream>>>(agg, x, gc1_rel_w,
                                                    gc1_root_w, gc1_rel_b, h1);
    gemm_ga_k<float><<<ga_blocks, 256, 0, stream>>>(h1, ga_W_w, ga_W_b, ga_U_w,
                                                    ga_U_b, ga_val_w, ga_val_b,
                                                    gated, value);
  }

  score_k<<<N_NODES, 256, 0, stream>>>(gated, ga_V_w, ga_V_b, scores);
  softmax_k<<<1, 256, 0, stream>>>(scores, alpha);
  zero_k<<<(int)(((long)N_GRAPHS * H1 + 255) / 256), 256, 0, stream>>>(
      pooled, (long)N_GRAPHS * H1);
  pool_k<<<N_NODES, 256, 0, stream>>>(value, alpha, batch, pooled);
  build_seq0_k<<<(SL * N_GRAPHS * 2 + 255) / 256, 256, 0, stream>>>(pooled,
                                                                    seq0);
  lstm_scan_k<<<1, 64, 0, stream>>>(seq0, 2, l0f_Wih, l0f_Whh, l0f_bih,
                                    l0f_bhh, seqA, 0, 0);
  lstm_scan_k<<<1, 64, 0, stream>>>(seq0, 2, l0b_Wih, l0b_Whh, l0b_bih,
                                    l0b_bhh, seqA, 3, 1);
  lstm_scan_k<<<1, 64, 0, stream>>>(seqA, 6, l1f_Wih, l1f_Whh, l1f_bih,
                                    l1f_bhh, seqB, 0, 0);
  lstm_scan_k<<<1, 64, 0, stream>>>(seqA, 6, l1b_Wih, l1b_Whh, l1b_bih,
                                    l1b_bhh, seqB, 3, 1);
  head_k<<<N_GRAPHS, 256, 0, stream>>>(seqB, mlp_w, mlp_b, (float*)d_out);
}
